// StructureAwareSparseAttention_8280696946914
// MI455X (gfx1250) — matrix operand order repaired and checked by reading, not compiled, not run on hardware
//
#include <hip/hip_runtime.h>
#include <hip/hip_bf16.h>

#define HIDDEN   512
#define HEADS    8
#define HEAD_DIM 64
#define SEQ_S    4096
#define SEQ_T    1024
#define MDIM     64
#define ATT_SCALE 0.125f   // HEAD_DIM^-0.5
#define EPS_F    1e-6f

typedef __bf16 bf16_t;
typedef __attribute__((ext_vector_type(16))) __bf16 v16bf;
typedef __attribute__((ext_vector_type(8)))  __bf16 v8bf;
typedef __attribute__((ext_vector_type(4)))  __bf16 v4bf;
typedef __attribute__((ext_vector_type(8)))  float   v8f;
typedef __attribute__((ext_vector_type(4)))  float   f4;

// GCC-vector int4 matching the async-LDS builtin's pointee type
typedef int i4v __attribute__((vector_size(16)));
typedef __attribute__((address_space(1))) i4v as1_i4;
typedef __attribute__((address_space(3))) i4v as3_i4;

#define WMMA_BF16(a, b, c) \
  __builtin_amdgcn_wmma_f32_16x16x32_bf16(false, (a), false, (b), (short)0, (c), false, false)

#if defined(__has_builtin)
#if __has_builtin(__builtin_amdgcn_global_load_async_to_lds_b128)
#define HAVE_ASYNC_LDS 1
#endif
#if __has_builtin(__builtin_amdgcn_s_wait_asynccnt)
#define WAIT_ASYNC(n) __builtin_amdgcn_s_wait_asynccnt(n)
#endif
#endif
#ifndef WAIT_ASYNC
#define WAIT_ASYNC(n) asm volatile("s_wait_asynccnt %0" ::"i"(n) : "memory")
#endif

// A-matrix 16x32 bf16 fragment: lane = row r; elems 0-7 -> K=hi*8+j, 8-15 -> K=16+hi*8+j
static __device__ __forceinline__ v16bf fragA_bf16(const bf16_t* __restrict__ p, int hi) {
  v8bf lo = *(const v8bf*)(p + hi * 8);
  v8bf hi8 = *(const v8bf*)(p + 16 + hi * 8);
  return __builtin_shufflevector(lo, hi8, 0, 1, 2, 3, 4, 5, 6, 7, 8, 9, 10, 11, 12, 13, 14, 15);
}

// B-matrix 32x16 bf16 fragment: lane = col r; elems 0-7 -> K=hi*8+j, 8-15 -> K=16+hi*8+j
static __device__ __forceinline__ v16bf fragB_bf16(const bf16_t* __restrict__ p, int hi) {
  v8bf lo = *(const v8bf*)(p + hi * 8);
  v8bf hi8 = *(const v8bf*)(p + 16 + hi * 8);
  return __builtin_shufflevector(lo, hi8, 0, 1, 2, 3, 4, 5, 6, 7, 8, 9, 10, 11, 12, 13, 14, 15);
}

// ---------------------------------------------------------------------------
// One-time fp32 -> bf16 conversion (removes all cvt VALU from the hot loops)
// ---------------------------------------------------------------------------
__global__ __launch_bounds__(256) void cvt_f32_bf16_kernel(const float* __restrict__ in,
                                                           bf16_t* __restrict__ out, int n4) {
  int i = blockIdx.x * 256 + threadIdx.x;
  if (i < n4) {
    f4 v = ((const f4*)in)[i];
    v4bf o;
#pragma unroll
    for (int j = 0; j < 4; ++j) o[j] = (__bf16)v[j];
    ((v4bf*)out)[i] = o;
  }
}

// ---------------------------------------------------------------------------
// GEMM: C[M,512] = A[M,512](bf16) @ W[512,512]^T(bf16) + bias
// MODE 0: bf16 row-major. MODE 1: bf16, V transposed -> [c,h,d,SEQ_T]. MODE 2: f32.
// One wave = one 16x16 tile; K-loop of 32; pure b128 loads, zero conversion.
// ---------------------------------------------------------------------------
template <int MODE>
__global__ __launch_bounds__(128) void gemm512_kernel(const bf16_t* __restrict__ A,
                                                      const bf16_t* __restrict__ W,
                                                      const float* __restrict__ bias,
                                                      void* __restrict__ outp) {
  int lane = threadIdx.x & 31;
  int w = threadIdx.x >> 5;
  int r = lane & 15, hi = lane >> 4;
  int m0 = blockIdx.x * 16;
  int n0 = (blockIdx.y * 4 + w) * 16;
  const bf16_t* arow = A + (size_t)(m0 + r) * HIDDEN;
  const bf16_t* wrow = W + (size_t)(n0 + r) * HIDDEN;
  v8f acc = {};
#pragma unroll 4
  for (int k0 = 0; k0 < HIDDEN; k0 += 32) {
    v16bf av = fragA_bf16(arow + k0, hi);
    v16bf bv = fragB_bf16(wrow + k0, hi);
    acc = WMMA_BF16(av, bv, acc);
  }
  float bval = bias[n0 + r];
#pragma unroll
  for (int g = 0; g < 8; ++g) {
    int row = m0 + g + 8 * hi;  // C/D layout: VGPR g + lane-half -> row; lane low4 -> col
    int col = n0 + r;
    float val = acc[g] + bval;
    if (MODE == 2) {
      ((float*)outp)[(size_t)row * HIDDEN + col] = val;
    } else if (MODE == 0) {
      ((bf16_t*)outp)[(size_t)row * HIDDEN + col] = (bf16_t)val;
    } else {  // V transposed: row = c*1024+t, col = h*64+d
      int cc = row >> 10, t = row & 1023;
      int hh = col >> 6, d = col & 63;
      ((bf16_t*)outp)[((size_t)((cc * HEADS + hh) * HEAD_DIM + d)) * SEQ_T + t] = (bf16_t)val;
    }
  }
}

// ---------------------------------------------------------------------------
// Async (double-buffered) staging of one 32x64 bf16 K chunk into LDS.
// 8 x b128 per wave; tracked on ASYNCcnt.
// ---------------------------------------------------------------------------
static __device__ __forceinline__ void stage_k_chunk(const bf16_t* __restrict__ gbase,
                                                     bf16_t* __restrict__ lbuf, int lane) {
#pragma unroll
  for (int i = 0; i < 8; ++i) {
    int idx = i * 32 + lane;          // 16-byte unit index within the 4KB tile
    int trow = idx >> 3;              // 0..31
    int d16 = idx & 7;                // which 8-elem group in the 64-wide row
    const bf16_t* g = gbase + trow * HIDDEN + d16 * 8;
    bf16_t* l = lbuf + trow * 64 + d16 * 8;
#ifdef HAVE_ASYNC_LDS
    __builtin_amdgcn_global_load_async_to_lds_b128((as1_i4*)g, (as3_i4*)l, 0, 0);
#else
    *(v8bf*)l = *(const v8bf*)g;      // sync fallback: global load + ds store (in-order)
#endif
  }
}

// ---------------------------------------------------------------------------
// Flash attention with fused structural bias.
// softmax(qk*scale + log p - log Z) == softmax(qk*scale + log p)  (shift-invariant,
// so the struct row normalization cancels). One wave per (c, h, 16-row q block).
// Softmax denominator kept as a WMMA accumulator against an all-ones B matrix
// (lane-local at the end); stabilization uses the running *tile* max, so the
// cross-lane reduction is 5 shuffles per chunk instead of 64.
// ---------------------------------------------------------------------------
__global__ __launch_bounds__(128) void attn_kernel(const bf16_t* __restrict__ Qb,
                                                   const bf16_t* __restrict__ Kb,
                                                   const bf16_t* __restrict__ Vt,
                                                   const bf16_t* __restrict__ Ub,
                                                   const bf16_t* __restrict__ Vsb,
                                                   bf16_t* __restrict__ AO) {
  __shared__ bf16_t ldsK[4][2][32 * 64];  // 32 KB: K double buffer per wave
  __shared__ bf16_t ldsP[4][512];         // 4 KB: P-transpose tile per wave
  int lane = threadIdx.x & 31;
  int w = threadIdx.x >> 5;
  int wid = blockIdx.x * 4 + w;
  int sblk = wid & 255;
  int h = (wid >> 8) & 7;
  int c = wid >> 11;
  int r = lane & 15, hi = lane >> 4;
  int s0 = sblk * 16;
  bf16_t* myP = ldsP[w];

  // persistent A-fragments: Q (K = head_dim 64) and struct-v (K = m 64)
  const bf16_t* qrow = Qb + ((size_t)(c * SEQ_S + s0 + r) * HIDDEN + h * HEAD_DIM);
  v16bf qa0 = fragA_bf16(qrow, hi);
  v16bf qa1 = fragA_bf16(qrow + 32, hi);
  const bf16_t* vrow = Vsb + (size_t)(c * SEQ_S + s0 + r) * MDIM;
  v16bf va0 = fragA_bf16(vrow, hi);
  v16bf va1 = fragA_bf16(vrow + 32, hi);

  v16bf onesB;  // all-ones B matrix for the row-sum accumulator
#pragma unroll
  for (int j = 0; j < 16; ++j) onesB[j] = (__bf16)1.0f;

  v8f acc[4] = {};
  v8f accl = {};          // softmax denominator accumulator (ones-column trick)
  float m_run = -1e30f;   // running tile max (valid upper bound for every row)

  const bf16_t* khead = Kb + ((size_t)c * SEQ_T * HIDDEN + h * HEAD_DIM);
  stage_k_chunk(khead, ldsK[w][0], lane);  // preload chunk 0

  for (int t0 = 0; t0 < SEQ_T; t0 += 32) {
    int cur = (t0 >> 5) & 1;
    if (t0 + 32 < SEQ_T) {
      stage_k_chunk(khead + (size_t)(t0 + 32) * HIDDEN, ldsK[w][cur ^ 1], lane);
      WAIT_ASYNC(8);  // async loads complete in order -> current buffer is ready
    } else {
      WAIT_ASYNC(0);
    }
    asm volatile("" ::: "memory");
    const bf16_t* kbuf = ldsK[w][cur];

    v8f sc[2];
#pragma unroll
    for (int tt = 0; tt < 2; ++tt) {
      // K B-fragments out of LDS: lane col r -> t-row 16*tt+r, contiguous d
      v16bf kb0 = fragB_bf16(kbuf + (16 * tt + r) * 64, hi);
      v16bf kb1 = fragB_bf16(kbuf + (16 * tt + r) * 64 + 32, hi);
      v8f s = {};
      s = WMMA_BF16(qa0, kb0, s);
      s = WMMA_BF16(qa1, kb1, s);
      const bf16_t* urow = Ub + (size_t)(c * SEQ_T + t0 + tt * 16 + r) * MDIM;
      v16bf ub0 = fragB_bf16(urow, hi);
      v16bf ub1 = fragB_bf16(urow + 32, hi);
      v8f st = {};
      st = WMMA_BF16(va0, ub0, st);
      st = WMMA_BF16(va1, ub1, st);
#pragma unroll
      for (int g = 0; g < 8; ++g)
        s[g] = s[g] * ATT_SCALE + __logf(fmaxf(st[g], EPS_F));
      sc[tt] = s;
    }
    if (t0 + 32 < SEQ_T)  // prefetch next V chunk (global_prefetch_b8)
      __builtin_prefetch(
          (const void*)(Vt + ((size_t)((c * HEADS + h) * HEAD_DIM + r) * SEQ_T + t0 + 32)), 0, 1);

    // tile-wide max: lane-local tree + 5-shuffle full-wave reduction
    float cm = sc[0][0];
#pragma unroll
    for (int g = 1; g < 8; ++g) cm = fmaxf(cm, sc[0][g]);
#pragma unroll
    for (int g = 0; g < 8; ++g) cm = fmaxf(cm, sc[1][g]);
    cm = fmaxf(cm, __shfl_xor(cm, 1));
    cm = fmaxf(cm, __shfl_xor(cm, 2));
    cm = fmaxf(cm, __shfl_xor(cm, 4));
    cm = fmaxf(cm, __shfl_xor(cm, 8));
    cm = fmaxf(cm, __shfl_xor(cm, 16));
    float mn = fmaxf(m_run, cm);
    float corr = __expf(m_run - mn);
    m_run = mn;
#pragma unroll
    for (int g = 0; g < 8; ++g) {
      sc[0][g] = __expf(sc[0][g] - mn);
      sc[1][g] = __expf(sc[1][g] - mn);
      accl[g] *= corr;
    }
#pragma unroll
    for (int jj = 0; jj < 4; ++jj)
#pragma unroll
      for (int g = 0; g < 8; ++g) acc[jj][g] *= corr;

    // transpose P (C layout) -> A-fragment layout via LDS; 8 rows/lane as b128
#pragma unroll
    for (int tt = 0; tt < 2; ++tt) {
      v8bf pv;
#pragma unroll
      for (int g = 0; g < 8; ++g) pv[g] = (__bf16)sc[tt][g];
      *(v8bf*)(myP + (r + 16 * tt) * 16 + 8 * hi) = pv;  // [k][m] layout
    }
    asm volatile("s_wait_dscnt 0" ::: "memory");
    v16bf pa;
#pragma unroll
    for (int j = 0; j < 8; ++j) {
      pa[j]     = myP[(hi * 8 + j) * 16 + r];
      pa[8 + j] = myP[(16 + hi * 8 + j) * 16 + r];
    }
    // P @ V (contiguous B-fragments thanks to transposed V) + row-sum column
#pragma unroll
    for (int jj = 0; jj < 4; ++jj) {
      const bf16_t* vtrow = Vt + ((size_t)((c * HEADS + h) * HEAD_DIM + jj * 16 + r) * SEQ_T + t0);
      v16bf vb = fragB_bf16(vtrow, hi);
      acc[jj] = WMMA_BF16(pa, vb, acc[jj]);
    }
    accl = WMMA_BF16(pa, onesB, accl);  // denominator: rowsum(P), lane-local
  }

#pragma unroll
  for (int jj = 0; jj < 4; ++jj)
#pragma unroll
    for (int g = 0; g < 8; ++g) {
      int row = s0 + g + 8 * hi;
      AO[(size_t)(c * SEQ_S + row) * HIDDEN + h * HEAD_DIM + jj * 16 + r] =
          (bf16_t)(acc[jj][g] / accl[g]);
    }
}

// ---------------------------------------------------------------------------
extern "C" void kernel_launch(void* const* d_in, const int* in_sizes, int n_in,
                              void* d_out, int out_size, void* d_ws, size_t ws_size,
                              hipStream_t stream) {
  (void)in_sizes; (void)n_in; (void)out_size; (void)ws_size;
  const float* x_tg = (const float*)d_in[0];
  const float* x_tf = (const float*)d_in[1];
  const float* u    = (const float*)d_in[2];
  const float* v    = (const float*)d_in[3];
  const float* Wq   = (const float*)d_in[4];
  const float* bq   = (const float*)d_in[5];
  const float* Wk   = (const float*)d_in[6];
  const float* bk   = (const float*)d_in[7];
  const float* Wv   = (const float*)d_in[8];
  const float* bv   = (const float*)d_in[9];
  const float* Wo   = (const float*)d_in[10];
  const float* bo   = (const float*)d_in[11];

  const size_t N_XTG = (size_t)4 * SEQ_S * HIDDEN;   // 8388608
  const size_t N_XTF = (size_t)4 * SEQ_T * HIDDEN;   // 2097152
  const size_t N_U   = (size_t)4 * SEQ_T * MDIM;     // 262144
  const size_t N_V   = (size_t)4 * SEQ_S * MDIM;     // 1048576
  const size_t N_W   = (size_t)HIDDEN * HIDDEN;      // 262144

  // Workspace layout (bf16 elements): ~68 MB total
  bf16_t* Qb  = (bf16_t*)d_ws;
  bf16_t* Kb  = Qb  + N_XTG;
  bf16_t* Vt  = Kb  + N_XTF;
  bf16_t* AO  = Vt  + N_XTF;
  bf16_t* Xtg = AO  + N_XTG;
  bf16_t* Xtf = Xtg + N_XTG;
  bf16_t* Ub  = Xtf + N_XTF;
  bf16_t* Vsb = Ub  + N_U;
  bf16_t* Wqb = Vsb + N_V;
  bf16_t* Wkb = Wqb + N_W;
  bf16_t* Wvb = Wkb + N_W;
  bf16_t* Wob = Wvb + N_W;

  // One-time bf16 conversions
  cvt_f32_bf16_kernel<<<(int)(N_XTG / 1024), 256, 0, stream>>>(x_tg, Xtg, (int)(N_XTG / 4));
  cvt_f32_bf16_kernel<<<(int)(N_XTF / 1024), 256, 0, stream>>>(x_tf, Xtf, (int)(N_XTF / 4));
  cvt_f32_bf16_kernel<<<(int)(N_U   / 1024), 256, 0, stream>>>(u,    Ub,  (int)(N_U   / 4));
  cvt_f32_bf16_kernel<<<(int)(N_V   / 1024), 256, 0, stream>>>(v,    Vsb, (int)(N_V   / 4));
  cvt_f32_bf16_kernel<<<(int)(N_W   / 1024), 256, 0, stream>>>(Wq,   Wqb, (int)(N_W   / 4));
  cvt_f32_bf16_kernel<<<(int)(N_W   / 1024), 256, 0, stream>>>(Wk,   Wkb, (int)(N_W   / 4));
  cvt_f32_bf16_kernel<<<(int)(N_W   / 1024), 256, 0, stream>>>(Wv,   Wvb, (int)(N_W   / 4));
  cvt_f32_bf16_kernel<<<(int)(N_W   / 1024), 256, 0, stream>>>(Wo,   Wob, (int)(N_W   / 4));

  // Projections (Q, K row-major; V transposed for contiguous PV B-fragments)
  gemm512_kernel<0><<<dim3(4 * SEQ_S / 16, 8), 128, 0, stream>>>(Xtg, Wqb, bq, Qb);
  gemm512_kernel<0><<<dim3(4 * SEQ_T / 16, 8), 128, 0, stream>>>(Xtf, Wkb, bk, Kb);
  gemm512_kernel<1><<<dim3(4 * SEQ_T / 16, 8), 128, 0, stream>>>(Xtf, Wvb, bv, Vt);

  // Flash attention with structural bias (4 waves / block)
  attn_kernel<<<dim3(4 * HEADS * (SEQ_S / 16) / 4), 128, 0, stream>>>(Qb, Kb, Vt, Ub, Vsb, AO);

  // Output projection -> fp32
  gemm512_kernel<2><<<dim3(4 * SEQ_S / 16, 8), 128, 0, stream>>>(AO, Wob, bo, (float*)d_out);
}
